// OptimalTransport_78056735638062
// MI455X (gfx1250) — matrix operand order repaired and checked by reading, not compiled, not run on hardware
//
#include <hip/hip_runtime.h>

// Sinkhorn OT on MI455X: K = exp(sim) resident in LDS as f16 (289.5 KB),
// per-iteration matvecs done on the WMMA pipe with the vector operand
// replicated across all 16 B-columns (D column 0 == full dot product).
// Grid: 768 blocks (one (b,h) slice each). Block: 384 threads = 12 wave32s.

#define W      384
#define NWAVE  12
#define STRD   193   // half2 (dword) stride per K-column; 193 % 64 == 1 -> conflict-free
#define ITERS  10

typedef _Float16 h2_t  __attribute__((ext_vector_type(2)));
typedef _Float16 v16h  __attribute__((ext_vector_type(16)));
typedef float    v8f   __attribute__((ext_vector_type(8)));

__device__ __forceinline__ float block_sum(float v, float* red, int lane, int wv) {
  #pragma unroll
  for (int m = 16; m; m >>= 1) v += __shfl_xor(v, m, 32);
  if (lane == 0) red[wv] = v;
  __syncthreads();
  float s = 0.f;
  #pragma unroll
  for (int k = 0; k < NWAVE; ++k) s += red[k];
  return s;  // uniform; later red writes are barrier-separated
}

__global__ __launch_bounds__(W, 1)
void sinkhorn_ot_wmma(const float* __restrict__ attn,
                      const float* __restrict__ phi_p,
                      float* __restrict__ out) {
  // tile[c*STRD + rp] = { K[c][2rp], K[c][2rp+1] },  K[c][r] = exp(attn[b,c,h,r])
  __shared__ h2_t  tile[W * STRD];          // 296,448 B
  __shared__ float eu[W + 1];               // exp(u), [W] = dustbin
  __shared__ float ev[W + 1];               // exp(v)
  __shared__ h2_t  euh[W / 2];              // f16 copy of eu body (B operand)
  __shared__ h2_t  evh[W / 2];              // f16 copy of ev body
  __shared__ float red[NWAVE];

  const int tid  = threadIdx.x;
  const int lane = tid & 31;
  const int wv   = tid >> 5;
  const int m16  = lane & 15;   // WMMA M index (A) / N index (unused, replicated B)
  const int hi   = lane >> 4;   // lane half-group
  const int bh   = blockIdx.x;
  const int b    = bh / W;
  const int h    = bh - b * W;

  const float phi = phi_p[0];
  const float P   = __expf(phi);          // dustbin kernel value
  const float MU  = 1.0f / (2.0f * W);    // body marginal 1/768 (mu == nu)

  // ---- Fill: one coalesced sweep of attn, K stored as f16 in LDS ----
  for (int i = 0; i < 32; ++i) {
    int c = wv * 32 + i;
    const float2* src = reinterpret_cast<const float2*>(
        attn + ((size_t)(b * W + c) * W + h) * (size_t)W);
    #pragma unroll
    for (int j = 0; j < 6; ++j) {
      float2 f = src[lane + 32 * j];
      h2_t kk;
      kk.x = (_Float16)__expf(f.x);
      kk.y = (_Float16)__expf(f.y);
      tile[c * STRD + lane + 32 * j] = kk;
    }
  }
  eu[tid] = 1.0f;
  if (tid == 0) { eu[W] = 1.0f; ev[W] = 1.0f; }
  __syncthreads();

  float SEU = (float)(W + 1);   // sum exp(u) with u == 0
  const _Float16* tileh = reinterpret_cast<const _Float16*>(tile);

  for (int it = 0; it < ITERS; ++it) {
    // ================= Phase A: ev[c] = MU / (K·eu + P*eu384) =================
    if (tid < W / 2) {
      h2_t p; p.x = (_Float16)eu[2 * tid]; p.y = (_Float16)eu[2 * tid + 1];
      euh[tid] = p;
    }
    __syncthreads();
    const float dustA = P * eu[W];

    v8f d0 = {}; v8f d1 = {};
    const int cb0 = (wv * 2) * 16, cb1 = cb0 + 16;   // two 16-column tiles per wave
    for (int j = 0; j < 12; ++j) {                   // K-chunks of 32 along r
      // B operand: eu chunk replicated over N. B map: K = 8*hi + 2v + h (v<4), 16 + 8*hi + 2(v-4) + h (v>=4)
      v16h bm;
      #pragma unroll
      for (int v = 0; v < 8; ++v) {
        h2_t p = euh[16 * j + 4 * hi + v + (v & 4)];
        bm[2 * v] = p.x; bm[2 * v + 1] = p.y;
      }
      // A operand: documented 16x32 f16 map; native r-major h2 pairs
      const h2_t* rowA0 = &tile[(cb0 + m16) * STRD + 16 * j];
      const h2_t* rowA1 = &tile[(cb1 + m16) * STRD + 16 * j];
      v16h a0, a1;
      #pragma unroll
      for (int v = 0; v < 4; ++v) {
        h2_t p0 = rowA0[4 * hi + v];     a0[2 * v] = p0.x;       a0[2 * v + 1] = p0.y;
        h2_t q0 = rowA0[8 + 4 * hi + v]; a0[2 * (v + 4)] = q0.x; a0[2 * (v + 4) + 1] = q0.y;
        h2_t p1 = rowA1[4 * hi + v];     a1[2 * v] = p1.x;       a1[2 * v + 1] = p1.y;
        h2_t q1 = rowA1[8 + 4 * hi + v]; a1[2 * (v + 4)] = q1.x; a1[2 * (v + 4) + 1] = q1.y;
      }
      d0 = __builtin_amdgcn_wmma_f32_16x16x32_f16(false, a0, false, bm, (short)0, d0, false, false);
      d1 = __builtin_amdgcn_wmma_f32_16x16x32_f16(false, a1, false, bm, (short)0, d1, false, false);
    }
    if (m16 == 0) {   // lanes 0 and 16 hold D column N=0 (M = j2 + 8*hi)
      #pragma unroll
      for (int j2 = 0; j2 < 8; ++j2) {
        ev[cb0 + j2 + 8 * hi] = MU / (d0[j2] + dustA);
        ev[cb1 + j2 + 8 * hi] = MU / (d1[j2] + dustA);
      }
    }
    __syncthreads();
    if (tid == 0) ev[W] = 0.5f / (P * SEU);    // dustbin column (closed form)
    __syncthreads();
    float SEV = block_sum(ev[tid] + (tid == 0 ? ev[W] : 0.f), red, lane, wv);

    // ================= Phase B: eu[r] = MU / (K^T·ev + P*ev384) ================
    if (tid < W / 2) {
      h2_t p; p.x = (_Float16)ev[2 * tid]; p.y = (_Float16)ev[2 * tid + 1];
      evh[tid] = p;
    }
    __syncthreads();
    const float dustB = P * ev[W];

    v8f e0 = {}; v8f e1 = {};
    const int rb0 = (wv * 2) * 16, rb1 = rb0 + 16;   // two 16-row tiles per wave
    for (int j = 0; j < 12; ++j) {                   // K-chunks of 32 along c
      v16h bm;
      #pragma unroll
      for (int v = 0; v < 8; ++v) {
        h2_t p = evh[16 * j + 4 * hi + v + (v & 4)];
        bm[2 * v] = p.x; bm[2 * v + 1] = p.y;
      }
      // A operand = K transposed: halves gathered with u16 reads (bank-cheap)
      v16h a0, a1;
      #pragma unroll
      for (int v = 0; v < 8; ++v) {
        int k0 = (v < 4) ? (2 * v + 8 * hi) : (16 + 2 * (v - 4) + 8 * hi);
        int c  = 32 * j + k0;
        a0[2 * v]     = tileh[(size_t)c       * (2 * STRD) + rb0 + m16];
        a0[2 * v + 1] = tileh[(size_t)(c + 1) * (2 * STRD) + rb0 + m16];
        a1[2 * v]     = tileh[(size_t)c       * (2 * STRD) + rb1 + m16];
        a1[2 * v + 1] = tileh[(size_t)(c + 1) * (2 * STRD) + rb1 + m16];
      }
      e0 = __builtin_amdgcn_wmma_f32_16x16x32_f16(false, a0, false, bm, (short)0, e0, false, false);
      e1 = __builtin_amdgcn_wmma_f32_16x16x32_f16(false, a1, false, bm, (short)0, e1, false, false);
    }
    if (m16 == 0) {
      #pragma unroll
      for (int j2 = 0; j2 < 8; ++j2) {
        eu[rb0 + j2 + 8 * hi] = MU / (e0[j2] + dustB);
        eu[rb1 + j2 + 8 * hi] = MU / (e1[j2] + dustB);
      }
    }
    __syncthreads();
    if (tid == 0) eu[W] = 0.5f / (P * SEV);    // dustbin row (closed form)
    __syncthreads();
    SEU = block_sum(eu[tid] + (tid == 0 ? eu[W] : 0.f), red, lane, wv);
  }

  // ---- Output: out[r][c] = K[c][r] * eu[r] * ev[c] * 768, coalesced in c ----
  float V[12];
  #pragma unroll
  for (int j = 0; j < 12; ++j) V[j] = ev[lane + 32 * j];
  float* outb = out + (size_t)bh * W * W;
  for (int p = 0; p < 16; ++p) {
    int rp = wv * 16 + p;                      // rows 2rp, 2rp+1
    float s0 = eu[2 * rp]     * (2.0f * W);
    float s1 = eu[2 * rp + 1] * (2.0f * W);
    #pragma unroll
    for (int j = 0; j < 12; ++j) {
      int c = lane + 32 * j;
      h2_t k = tile[c * STRD + rp];
      outb[(size_t)(2 * rp) * W + c]     = (float)k.x * s0 * V[j];
      outb[(size_t)(2 * rp + 1) * W + c] = (float)k.y * s1 * V[j];
    }
  }
}

extern "C" void kernel_launch(void* const* d_in, const int* in_sizes, int n_in,
                              void* d_out, int out_size, void* d_ws, size_t ws_size,
                              hipStream_t stream) {
  (void)in_sizes; (void)n_in; (void)d_ws; (void)ws_size; (void)out_size;
  const float* attn = (const float*)d_in[0];
  const float* phi  = (const float*)d_in[1];
  float* out = (float*)d_out;
  sinkhorn_ot_wmma<<<dim3(2 * W), dim3(W), 0, stream>>>(attn, phi, out);
}
